// Block_36455682408804
// MI455X (gfx1250) — matrix operand order repaired and checked by reading, not compiled, not run on hardware
//
#include <hip/hip_runtime.h>
#include <hip/hip_bf16.h>
#include <math.h>

// ---------------------------------------------------------------------------
// Transformer block for MI455X (gfx1250): bf16 WMMA, fp32 accumulate.
// B=16 T=1024 D=128 H=8 HS=16.  All WMMA operands built from b128 loads.
// ---------------------------------------------------------------------------

typedef __bf16 bf16;
typedef __attribute__((ext_vector_type(16))) __bf16 v16bf;
typedef __attribute__((ext_vector_type(8)))  __bf16 v8bf;
typedef __attribute__((ext_vector_type(8)))  float  v8f;

#define NB   16
#define NT   1024
#define ND   128
#define NH   8
#define NHS  16
#define NBT  (NB * NT)        // 16384 rows
#define NFF  (4 * ND)         // 512
#define NQKV (3 * ND)         // 384 fused q|k|v columns

__device__ inline v8f wmma_bf16(v16bf a, v16bf b, v8f c) {
  return __builtin_amdgcn_wmma_f32_16x16x32_bf16(
      false, a, false, b, (short)0, c, false, false);
}

__device__ inline v16bf cat8(v8bf lo, v8bf hi) {
  v16bf r = __builtin_shufflevector(lo, hi, 0, 1, 2, 3, 4, 5, 6, 7,
                                    8, 9, 10, 11, 12, 13, 14, 15);
  return r;
}

// ---- operand loaders (layouts per CDNA5 ISA §7.12.2) ------------
// A (16x32): row = lane&15; elems 0..7 -> K = 8*(lane>>4)+j, elems 8..15 ->
// K = 16 + 8*(lane>>4) + j.  Row-major source, two 16B loads.
__device__ inline v16bf ld_a32(const bf16* p, int ld) {
  int lane = threadIdx.x & 31;
  const bf16* row = p + (size_t)(lane & 15) * ld + (lane >> 4) * 8;
  v8bf a0 = *(const v8bf*)row;
  v8bf a1 = *(const v8bf*)(row + 16);
  return cat8(a0, a1);
}

// A with only K=0..15 valid (HS=16), upper K zero-padded.
__device__ inline v16bf ld_a16(const bf16* p, int ld) {
  int lane = threadIdx.x & 31;
  const bf16* row = p + (size_t)(lane & 15) * ld + (lane >> 4) * 8;
  v8bf a0 = *(const v8bf*)row;
  v8bf a1 = {};
  return cat8(a0, a1);
}

// B (32x16): col N = lane&15, elems 0..7 -> K = 8*(lane>>4)+j, elems 8..15 ->
// K = 16 + 8*(lane>>4) + j.  Source is K-contiguous
// per column (transposed weights, row stride ldk) -> two 16B loads.
__device__ inline v16bf ld_b32(const bf16* pT, int ldk) {
  int lane = threadIdx.x & 31;
  const bf16* row = pT + (size_t)(lane & 15) * ldk + (lane >> 4) * 8;
  v8bf b0 = *(const v8bf*)row;
  v8bf b1 = *(const v8bf*)(row + 16);
  return cat8(b0, b1);
}

// B with only K=0..15 valid: elems 0..7 -> K = 8*(lane>>4)+j, elems 8..15 zero.
__device__ inline v16bf ld_b16(const bf16* pT, int ldk) {
  int lane = threadIdx.x & 31;
  const bf16* row = pT + (size_t)(lane & 15) * ldk + (lane >> 4) * 8;
  v8bf z = {};
  v8bf b0 = *(const v8bf*)row;
  v8bf b1 = z;
  return cat8(b0, b1);
}

// ---------------------------------------------------------------------------
__global__ void cvt_f32_bf16(const float* __restrict__ s, bf16* __restrict__ d, int n) {
  int i = blockIdx.x * 256 + threadIdx.x;
  if (i < n) d[i] = (bf16)s[i];
}

// transpose-convert: s[K,N] f32 -> d[N,K] bf16
__global__ void cvt_t_bf16(const float* __restrict__ s, bf16* __restrict__ d, int K, int N) {
  int i = blockIdx.x * 256 + threadIdx.x;
  if (i < K * N) {
    int k = i / N, n = i % N;
    d[(size_t)n * K + k] = (bf16)s[i];
  }
}

// fuse per-head Wq/Wk/Wv [H,D,HS] into transposed panel WqkvT[384,128]
__global__ void cvt_qkv_w(const float* __restrict__ Wq, const float* __restrict__ Wk,
                          const float* __restrict__ Wv, bf16* __restrict__ WqkvT) {
  int i = blockIdx.x * 256 + threadIdx.x;        // 3*8*128*16 = 49152
  if (i >= 3 * NH * ND * NHS) return;
  int which = i / (NH * ND * NHS);
  int r = i % (NH * ND * NHS);
  int h = r / (ND * NHS);
  int k = (r / NHS) % ND;
  int s = r % NHS;
  const float* W = (which == 0) ? Wq : (which == 1) ? Wk : Wv;
  int n = which * ND + h * NHS + s;              // fused column 0..383
  WqkvT[(size_t)n * ND + k] = (bf16)W[(size_t)h * ND * NHS + k * NHS + s];
}

// ---- fused QKV GEMM: xb[16384,128] x WqkvT -> q,k [B,H,T,16], v^T [B,H,16,T]
__global__ __launch_bounds__(256) void qkv_kernel(
    const bf16* __restrict__ xb, const bf16* __restrict__ WqkvT,
    const float* __restrict__ bq, const float* __restrict__ bk, const float* __restrict__ bv,
    bf16* __restrict__ qb, bf16* __restrict__ kb, bf16* __restrict__ vtb) {
  int lane = threadIdx.x & 31;
  int gw = blockIdx.x * 8 + (threadIdx.x >> 5);   // 6144 waves
  int cg = gw % 6;                                // 64-col group (384/64)
  int tile = gw / 6;                              // BT row tile 0..1023
  const bf16* arow = xb + (size_t)tile * 16 * ND;
  const bf16* wbase = WqkvT + (size_t)cg * 64 * ND;
  __builtin_prefetch(wbase, 0, 0);

  v8f acc[4] = {};
#pragma unroll
  for (int kt = 0; kt < 4; ++kt) {
    v16bf a = ld_a32(arow + kt * 32, ND);
#pragma unroll
    for (int j = 0; j < 4; ++j) {
      v16bf b = ld_b32(wbase + (size_t)(j * 16) * ND + kt * 32, ND);
      acc[j] = wmma_bf16(a, b, acc[j]);
    }
  }

  int col = lane & 15;
  int ho = (lane >> 4) * 8;
#pragma unroll
  for (int j = 0; j < 4; ++j) {
    int n = cg * 64 + j * 16 + col;               // fused column
    int which = n >> 7;
    int h = (n >> 4) & 7;
    int s = n & 15;
    const float* bias = (which == 0) ? bq : (which == 1) ? bk : bv;
    float bz = bias[h * NHS + s];
#pragma unroll
    for (int i = 0; i < 8; ++i) {
      int r = tile * 16 + i + ho;
      int bidx = r >> 10;
      int t = r & (NT - 1);
      bf16 val = (bf16)(acc[j][i] + bz);
      if (which == 2)
        vtb[(((size_t)bidx * NH + h) * NHS + s) * NT + t] = val;   // V transposed
      else {
        bf16* dst = which ? kb : qb;
        dst[(((size_t)bidx * NH + h) * NT + t) * NHS + s] = val;
      }
    }
  }
}

// ---- Flash attention: one wave per (b, h, 16-row q tile) -------------------
__global__ __launch_bounds__(256) void attn_kernel(
    const bf16* __restrict__ qb, const bf16* __restrict__ kb,
    const bf16* __restrict__ vtb, bf16* __restrict__ attnb) {
  __shared__ __align__(32) bf16 stage[8][16][16];
  int lane = threadIdx.x & 31;
  int wv = threadIdx.x >> 5;
  int gw = blockIdx.x * 8 + wv;                   // 8192 waves
  int b = gw >> 9;
  int h = (gw >> 6) & 7;
  int tq = gw & 63;

  const bf16* qh = qb + ((size_t)(b * NH + h) * NT) * NHS;
  const bf16* kh = kb + ((size_t)(b * NH + h) * NT) * NHS;
  const bf16* vth = vtb + ((size_t)(b * NH + h) * NHS) * NT;

  v16bf aq = ld_a16(qh + (size_t)tq * 16 * NHS, NHS);

  int col = lane & 15;
  int ho = (lane >> 4) * 8;

  float m8[8], l8[8];
  v8f acc = {};
#pragma unroll
  for (int i = 0; i < 8; ++i) { m8[i] = -3.0e38f; l8[i] = 0.0f; }

  for (int ku = 0; ku <= tq; ++ku) {
    // B = k^T: col u of B is row u of k (contiguous 16 bf16)
    v16bf bk16 = ld_b16(kh + (size_t)ku * 16 * NHS, NHS);
    v8f cz = {};
    v8f s = wmma_bf16(aq, bk16, cz);

#pragma unroll
    for (int i = 0; i < 8; ++i) {
      float sv = s[i] * 0.25f;                    // 1/sqrt(HS)
      if (ku == tq && col > (i + ho)) sv = -1.0e30f;   // causal mask
      s[i] = sv;
    }

#pragma unroll
    for (int i = 0; i < 8; ++i) {
      float red = s[i];                           // row max over 16-lane half
      red = fmaxf(red, __shfl_xor(red, 1));
      red = fmaxf(red, __shfl_xor(red, 2));
      red = fmaxf(red, __shfl_xor(red, 4));
      red = fmaxf(red, __shfl_xor(red, 8));
      float mn = fmaxf(m8[i], red);
      float sc = __expf(m8[i] - mn);
      float p  = __expf(s[i] - mn);
      float rs = p;
      rs += __shfl_xor(rs, 1);
      rs += __shfl_xor(rs, 2);
      rs += __shfl_xor(rs, 4);
      rs += __shfl_xor(rs, 8);
      l8[i] = l8[i] * sc + rs;
      acc[i] *= sc;
      m8[i] = mn;
      stage[wv][i + ho][col] = (bf16)p;           // P (C layout) -> LDS
    }
    asm volatile("" ::: "memory");
    v16bf ap = ld_a16(&stage[wv][0][0], 16);      // P re-laid-out as A operand
    // B = V tile: col s of B is row s of V^T (contiguous 16 bf16)
    v16bf bv16 = ld_b16(vth + (size_t)ku * 16, NT);
    acc = wmma_bf16(ap, bv16, acc);
    asm volatile("" ::: "memory");
  }

  bf16* orow = attnb + (size_t)b * NT * ND + h * NHS;
#pragma unroll
  for (int i = 0; i < 8; ++i) {
    int t = tq * 16 + i + ho;
    orow[(size_t)t * ND + col] = (bf16)(acc[i] / l8[i]);
  }
}

// ---- attn @ Wp + bp + x, then LayerNorm1 (64 rows per block) ---------------
__global__ __launch_bounds__(256) void proj_ln1_kernel(
    const bf16* __restrict__ attnb, const bf16* __restrict__ WpT,
    const float* __restrict__ bp, const float* __restrict__ x,
    const float* __restrict__ g1, const float* __restrict__ be1,
    float* __restrict__ ln1f, bf16* __restrict__ ln1b) {
  __shared__ float srow[64][ND];                  // 32 KB
  int lane = threadIdx.x & 31;
  int wv = threadIdx.x >> 5;
  int rs = wv >> 1;                               // row subtile 0..3
  int ch = wv & 1;                                // 64-col half
  size_t rbase = (size_t)blockIdx.x * 64;
  const bf16* arow = attnb + (rbase + rs * 16) * ND;
  const bf16* wbase = WpT + (size_t)ch * 64 * ND;
  __builtin_prefetch(wbase, 0, 0);

  v8f acc[4] = {};
#pragma unroll
  for (int kt = 0; kt < 4; ++kt) {
    v16bf a = ld_a32(arow + kt * 32, ND);
#pragma unroll
    for (int j = 0; j < 4; ++j) {
      v16bf b = ld_b32(wbase + (size_t)(j * 16) * ND + kt * 32, ND);
      acc[j] = wmma_bf16(a, b, acc[j]);
    }
  }

  int col = lane & 15;
  int ho = (lane >> 4) * 8;
#pragma unroll
  for (int j = 0; j < 4; ++j) {
    int c = ch * 64 + j * 16 + col;
#pragma unroll
    for (int i = 0; i < 8; ++i) {
      int row = rs * 16 + i + ho;
      srow[row][c] = acc[j][i] + bp[c] + x[(rbase + row) * ND + c];
    }
  }
  __syncthreads();

  if (threadIdx.x < 64) {
    int row = threadIdx.x;
    float mu = 0.0f;
    for (int j2 = 0; j2 < ND; ++j2) mu += srow[row][j2];
    mu *= (1.0f / ND);
    float var = 0.0f;
    for (int j2 = 0; j2 < ND; ++j2) { float d = srow[row][j2] - mu; var += d * d; }
    var *= (1.0f / ND);
    float rstd = rsqrtf(var + 1e-5f);
    size_t base = (rbase + row) * ND;
    for (int j2 = 0; j2 < ND; ++j2) {
      float y = (srow[row][j2] - mu) * rstd * g1[j2] + be1[j2];
      ln1f[base + j2] = y;
      ln1b[base + j2] = (bf16)y;
    }
  }
}

// ---- MLP fc1 + exact GELU (16x64 per wave) ---------------------------------
__global__ __launch_bounds__(256) void mlp1_kernel(
    const bf16* __restrict__ ln1b, const bf16* __restrict__ W1T,
    const float* __restrict__ b1, bf16* __restrict__ h1b) {
  int lane = threadIdx.x & 31;
  int gw = blockIdx.x * 8 + (threadIdx.x >> 5);   // 8192 waves
  int cg = gw & 7;                                // 64-col group (512/64)
  size_t rt = (size_t)(gw >> 3);                  // row tile 0..1023
  const bf16* arow = ln1b + rt * 16 * ND;
  const bf16* wbase = W1T + (size_t)cg * 64 * ND;
  __builtin_prefetch(wbase, 0, 0);

  v8f acc[4] = {};
#pragma unroll
  for (int kt = 0; kt < 4; ++kt) {
    v16bf a = ld_a32(arow + kt * 32, ND);
#pragma unroll
    for (int j = 0; j < 4; ++j) {
      v16bf b = ld_b32(wbase + (size_t)(j * 16) * ND + kt * 32, ND);
      acc[j] = wmma_bf16(a, b, acc[j]);
    }
  }

  int col = lane & 15;
  int ho = (lane >> 4) * 8;
#pragma unroll
  for (int j = 0; j < 4; ++j) {
    int c = cg * 64 + j * 16 + col;
    float bias = b1[c];
#pragma unroll
    for (int i = 0; i < 8; ++i) {
      float z = acc[j][i] + bias;
      float g = 0.5f * z * (1.0f + erff(z * 0.70710678118654752f));  // exact GELU
      h1b[(rt * 16 + i + ho) * NFF + c] = (bf16)g;
    }
  }
}

// ---- MLP fc2 + residual + LayerNorm2 -> d_out (64 rows per block) ----------
__global__ __launch_bounds__(256) void mlp2_ln2_kernel(
    const bf16* __restrict__ h1b, const bf16* __restrict__ W2T,
    const float* __restrict__ b2, const float* __restrict__ ln1f,
    const float* __restrict__ g2, const float* __restrict__ be2,
    float* __restrict__ out) {
  __shared__ float srow[64][ND];                  // 32 KB
  int lane = threadIdx.x & 31;
  int wv = threadIdx.x >> 5;
  int rs = wv >> 1;
  int ch = wv & 1;
  size_t rbase = (size_t)blockIdx.x * 64;
  const bf16* arow = h1b + (rbase + rs * 16) * NFF;
  const bf16* wbase = W2T + (size_t)ch * 64 * NFF;
  __builtin_prefetch(wbase, 0, 0);

  v8f acc[4] = {};
#pragma unroll
  for (int kt = 0; kt < 16; ++kt) {
    v16bf a = ld_a32(arow + kt * 32, NFF);
#pragma unroll
    for (int j = 0; j < 4; ++j) {
      v16bf b = ld_b32(wbase + (size_t)(j * 16) * NFF + kt * 32, NFF);
      acc[j] = wmma_bf16(a, b, acc[j]);
    }
  }

  int col = lane & 15;
  int ho = (lane >> 4) * 8;
#pragma unroll
  for (int j = 0; j < 4; ++j) {
    int c = ch * 64 + j * 16 + col;
#pragma unroll
    for (int i = 0; i < 8; ++i) {
      int row = rs * 16 + i + ho;
      srow[row][c] = acc[j][i] + b2[c] + ln1f[(rbase + row) * ND + c];
    }
  }
  __syncthreads();

  if (threadIdx.x < 64) {
    int row = threadIdx.x;
    float mu = 0.0f;
    for (int j2 = 0; j2 < ND; ++j2) mu += srow[row][j2];
    mu *= (1.0f / ND);
    float var = 0.0f;
    for (int j2 = 0; j2 < ND; ++j2) { float d = srow[row][j2] - mu; var += d * d; }
    var *= (1.0f / ND);
    float rstd = rsqrtf(var + 1e-5f);
    size_t base = (rbase + row) * ND;
    for (int j2 = 0; j2 < ND; ++j2)
      out[base + j2] = (srow[row][j2] - mu) * rstd * g2[j2] + be2[j2];
  }
}

// ---------------------------------------------------------------------------
extern "C" void kernel_launch(void* const* d_in, const int* in_sizes, int n_in,
                              void* d_out, int out_size, void* d_ws, size_t ws_size,
                              hipStream_t stream) {
  const float* x   = (const float*)d_in[0];
  const float* Wq  = (const float*)d_in[1];
  const float* bq  = (const float*)d_in[2];
  const float* Wk  = (const float*)d_in[3];
  const float* bk  = (const float*)d_in[4];
  const float* Wv  = (const float*)d_in[5];
  const float* bv  = (const float*)d_in[6];
  const float* Wp  = (const float*)d_in[7];
  const float* bp  = (const float*)d_in[8];
  const float* W1  = (const float*)d_in[9];
  const float* b1  = (const float*)d_in[10];
  const float* W2  = (const float*)d_in[11];
  const float* b2  = (const float*)d_in[12];
  const float* g1  = (const float*)d_in[13];
  const float* be1 = (const float*)d_in[14];
  const float* g2  = (const float*)d_in[15];
  const float* be2 = (const float*)d_in[16];

  char* w = (char*)d_ws;
  auto carve = [&](size_t bytes) -> char* {
    char* p = w;
    w += (bytes + 255) & ~(size_t)255;
    return p;
  };

  bf16* xb    = (bf16*)carve((size_t)NBT * ND * 2);           // 4 MB
  bf16* WqkvT = (bf16*)carve((size_t)NQKV * ND * 2);          // 96 KB
  bf16* WpT   = (bf16*)carve((size_t)ND * ND * 2);            // 32 KB
  bf16* W1T   = (bf16*)carve((size_t)NFF * ND * 2);           // 128 KB (transposed [512,128])
  bf16* W2T   = (bf16*)carve((size_t)ND * NFF * 2);           // 128 KB (transposed [128,512])
  bf16* qb    = (bf16*)carve((size_t)NB * NH * NT * NHS * 2); // 4 MB
  bf16* kb    = (bf16*)carve((size_t)NB * NH * NT * NHS * 2);
  bf16* vtb   = (bf16*)carve((size_t)NB * NH * NHS * NT * 2);
  bf16* attnb = (bf16*)carve((size_t)NBT * ND * 2);           // 4 MB
  float* ln1f = (float*)carve((size_t)NBT * ND * 4);          // 8 MB
  bf16* ln1b  = (bf16*)carve((size_t)NBT * ND * 2);           // 4 MB
  bf16* h1b   = (bf16*)carve((size_t)NBT * NFF * 2);          // 16 MB

  // conversions / transposes
  cvt_f32_bf16<<<(NBT * ND + 255) / 256, 256, 0, stream>>>(x, xb, NBT * ND);
  cvt_qkv_w<<<(3 * NH * ND * NHS + 255) / 256, 256, 0, stream>>>(Wq, Wk, Wv, WqkvT);
  cvt_t_bf16<<<(ND * ND + 255) / 256, 256, 0, stream>>>(Wp, WpT, ND, ND);
  cvt_t_bf16<<<(ND * NFF + 255) / 256, 256, 0, stream>>>(W1, W1T, ND, NFF);
  cvt_t_bf16<<<(NFF * ND + 255) / 256, 256, 0, stream>>>(W2, W2T, NFF, ND);

  // fused QKV: 1024 row tiles * 6 col groups = 6144 waves / 8 per block
  qkv_kernel<<<768, 256, 0, stream>>>(xb, WqkvT, bq, bk, bv, qb, kb, vtb);
  // 16 * 8 * 64 = 8192 waves
  attn_kernel<<<1024, 256, 0, stream>>>(qb, kb, vtb, attnb);
  // 16384 / 64 rows per block
  proj_ln1_kernel<<<256, 256, 0, stream>>>(attnb, WpT, bp, x, g1, be1, ln1f, ln1b);
  // 1024 row tiles * 8 col groups = 8192 waves
  mlp1_kernel<<<1024, 256, 0, stream>>>(ln1b, W1T, b1, h1b);
  mlp2_ln2_kernel<<<256, 256, 0, stream>>>(h1b, W2T, b2, ln1f, g2, be2, (float*)d_out);
}
